// NewtonNet_46548855554170
// MI455X (gfx1250) — matrix operand order repaired and checked by reading, not compiled, not run on hardware
//
#include <hip/hip_runtime.h>
#include <hip/hip_bf16.h>

#define F 128
#define NB 20
#define LAYERS 3
#define RC 5.0f
#define PI_F 3.14159265358979323846f

typedef __attribute__((ext_vector_type(16))) _Float16 v16h;
typedef __attribute__((ext_vector_type(8)))  float    v8f;

// ---------------------------------------------------------------------------
// WMMA fragment layouts (v_wmma_f32_16x16x32_f16, wave32)
// A (16x32 f16): lanes 0-15 row M=lane, K = kb..kb+7 (v0-3) and kb+16..kb+23
//                (v4-7); lanes 16-31 same rows, K shifted by +8.
// B (32x16 f16): lane holds column N=lane&15; lanes 0-15 K=kb..kb+7, kb+16..kb+23,
//                lanes 16-31 K shifted by +8.  Every GEMM here is X @ W^T,
//                so B column n == W row n  =>  two contiguous 16B loads.
// C/D (16x16 f32): lane&15 = N, vgpr i -> M = i + 8*(lane>>4).
// ---------------------------------------------------------------------------

__device__ __forceinline__ v16h load_a_lds(const _Float16* h, int lane, int kb) {
  const int r = lane & 15;
  const int k = kb + ((lane >> 4) << 3);           // +8 for upper half-wave
  const _Float16* p = h + r * F + k;
  v16h a;
#pragma unroll
  for (int j = 0; j < 8; ++j) {
    a[j]     = p[j];
    a[8 + j] = p[16 + j];
  }
  return a;
}

__device__ __forceinline__ v16h load_b_wt(const _Float16* __restrict__ W,
                                          int col0, int lane, int kb) {
  const int c = col0 + (lane & 15);
  const int k = kb + ((lane >> 4) << 3);           // +8 for upper half-wave
  const _Float16* p = W + (size_t)c * F + k;
  v16h b;
#pragma unroll
  for (int j = 0; j < 8; ++j) {
    b[j]     = p[j];
    b[8 + j] = p[16 + j];
  }
  return b;
}

// Cooperative A-tile stage: 256 threads bring 16 rows x 128 cols of X into
// LDS as f16 (one pass; all 8 waves then feed fragments from ds_read).
template <typename T>
__device__ __forceinline__ void stage_a_tile(const T* __restrict__ X, int row0,
                                             _Float16* abuf) {
  const int t = threadIdx.x;
  const int r = t >> 4;                 // 0..15
  const int c = (t & 15) << 3;          // 0,8,...,120
  const T* src = X + (size_t)(row0 + r) * F + c;
  _Float16* dst = abuf + r * F + c;
#pragma unroll
  for (int j = 0; j < 8; ++j) dst[j] = (_Float16)src[j];
}

__device__ __forceinline__ float silu(float x) {
  return x / (1.f + __expf(-x));
}

// ---------------------------------------------------------------------------
// Fused 2-layer MLP:  Y = silu(X @ W1^T + b1) @ W2^T + b2
// A tile and hidden tile both staged through LDS (8 KB total).
// Block = 256 threads = 8 waves -> one 16-row x 128-col tile.
// M must be a multiple of 16 (20000 / 320000: exact).
// ---------------------------------------------------------------------------
template <typename T>
__global__ void __launch_bounds__(256)
mlp2_wmma_kernel(const T* __restrict__ X,
                 const _Float16* __restrict__ W1, const float* __restrict__ b1,
                 const _Float16* __restrict__ W2, const float* __restrict__ b2,
                 float* __restrict__ Y) {
  __shared__ __align__(16) _Float16 abuf[16 * F];   // 4 KB input tile
  __shared__ __align__(16) _Float16 hbuf[16 * F];   // 4 KB hidden tile

  const int wave = threadIdx.x >> 5;
  const int lane = threadIdx.x & 31;
  const int row0 = blockIdx.x << 4;
  const int col  = (wave << 4) + (lane & 15);
  const int mofs = (lane >> 4) << 3;                // 0 or 8

  stage_a_tile(X, row0, abuf);
  __syncthreads();

  // ---- stage 1: h = silu(A @ W1^T + b1) ----
  v8f acc = {};
#pragma unroll
  for (int kb = 0; kb < F; kb += 32) {
    v16h a = load_a_lds(abuf, lane, kb);
    v16h b = load_b_wt(W1, wave << 4, lane, kb);
    if (kb + 32 < F)
      __builtin_prefetch(W1 + (size_t)col * F + kb + 32, 0, 1);
    acc = __builtin_amdgcn_wmma_f32_16x16x32_f16(false, a, false, b,
                                                 (short)0, acc, false, false);
  }
  const float bias1 = b1 ? b1[col] : 0.f;
#pragma unroll
  for (int i = 0; i < 8; ++i)
    hbuf[(mofs + i) * F + col] = (_Float16)silu(acc[i] + bias1);
  __syncthreads();

  // ---- stage 2: Y = h @ W2^T + b2 ----
  v8f acc2 = {};
#pragma unroll
  for (int kb = 0; kb < F; kb += 32) {
    v16h a = load_a_lds(hbuf, lane, kb);
    v16h b = load_b_wt(W2, wave << 4, lane, kb);
    acc2 = __builtin_amdgcn_wmma_f32_16x16x32_f16(false, a, false, b,
                                                  (short)0, acc2, false, false);
  }
  const float bias2 = b2 ? b2[col] : 0.f;
  float* Yp = Y + (size_t)row0 * F + col;
#pragma unroll
  for (int i = 0; i < 8; ++i)
    Yp[(size_t)(mofs + i) * F] = acc2[i] + bias2;
}

// ---------------------------------------------------------------------------
// Single GEMM:  Y = X @ W^T   (fp32 X; used for force @ upd_W^T, M = 3N)
// ---------------------------------------------------------------------------
__global__ void __launch_bounds__(256)
gemm_wmma_kernel(const float* __restrict__ X, const _Float16* __restrict__ W,
                 float* __restrict__ Y) {
  __shared__ __align__(16) _Float16 abuf[16 * F];

  const int wave = threadIdx.x >> 5;
  const int lane = threadIdx.x & 31;
  const int row0 = blockIdx.x << 4;
  const int col  = (wave << 4) + (lane & 15);
  const int mofs = (lane >> 4) << 3;

  stage_a_tile(X, row0, abuf);
  __syncthreads();

  v8f acc = {};
#pragma unroll
  for (int kb = 0; kb < F; kb += 32) {
    v16h a = load_a_lds(abuf, lane, kb);
    v16h b = load_b_wt(W, wave << 4, lane, kb);
    acc = __builtin_amdgcn_wmma_f32_16x16x32_f16(false, a, false, b,
                                                 (short)0, acc, false, false);
  }
  float* Yp = Y + (size_t)row0 * F + col;
#pragma unroll
  for (int i = 0; i < 8; ++i)
    Yp[(size_t)(mofs + i) * F] = acc[i];
}

// ---------------------------------------------------------------------------
// Elementwise / gather-scatter kernels (memory-bound path)
// ---------------------------------------------------------------------------
__global__ void f32_to_f16_kernel(const float* __restrict__ s,
                                  _Float16* __restrict__ d, int n) {
  int i = blockIdx.x * blockDim.x + threadIdx.x;
  if (i < n) d[i] = (_Float16)s[i];
}

__global__ void __launch_bounds__(128)
init_kernel(const int* __restrict__ z, const float* __restrict__ emb,
            float* __restrict__ atom, float* __restrict__ force) {
  const int n = blockIdx.x, f = threadIdx.x;
  atom[(size_t)n * F + f] = emb[(size_t)z[n] * F + f];
#pragma unroll
  for (int v = 0; v < 3; ++v)
    force[((size_t)n * 3 + v) * F + f] = 0.f;
}

__global__ void edge_embed_kernel(const float* __restrict__ disp,
                                  float* __restrict__ dist_edge,
                                  float* __restrict__ dir_edge, int E) {
  int e = blockIdx.x * blockDim.x + threadIdx.x;
  if (e >= E) return;
  float dx = disp[e * 3], dy = disp[e * 3 + 1], dz = disp[e * 3 + 2];
  float d = sqrtf(dx * dx + dy * dy + dz * dz);
  float inv = 1.f / d;
  dir_edge[e * 3]     = dx * inv;
  dir_edge[e * 3 + 1] = dy * inv;
  dir_edge[e * 3 + 2] = dz * inv;
  float cut = (d < RC) ? 0.5f * (__cosf(PI_F * d * (1.f / RC)) + 1.f) : 0.f;
  float pref = sqrtf(2.f / RC) * inv * cut;
  float w = PI_F * d * (1.f / RC);
#pragma unroll
  for (int b = 0; b < NB; ++b)
    dist_edge[(size_t)e * NB + b] = pref * __sinf((float)(b + 1) * w);
}

// msg = (dist_edge @ me_W^T) * mnp[ei0] * mnp[ei1] ; atom[ei0] += msg
// msg stored as f16 (only ever consumed as a WMMA A operand).
__global__ void __launch_bounds__(128)
msg_kernel(const float* __restrict__ dist_edge, const float* __restrict__ meW,
           const float* __restrict__ mnp, const int* __restrict__ ei0,
           const int* __restrict__ ei1, _Float16* __restrict__ msg,
           float* __restrict__ atom) {
  const int e = blockIdx.x, f = threadIdx.x;
  __shared__ float de[NB];
  if (f < NB) de[f] = dist_edge[(size_t)e * NB + f];
  __syncthreads();
  float mep = 0.f;
#pragma unroll
  for (int b = 0; b < NB; ++b) mep += de[b] * meW[f * NB + b];
  const int i0 = ei0[e], i1 = ei1[e];
  float v = mep * mnp[(size_t)i0 * F + f] * mnp[(size_t)i1 * F + f];
  msg[(size_t)e * F + f] = (_Float16)v;
  atomicAdd(&atom[(size_t)i0 * F + f], v);
}

// force_out[ei0] += t1*dir + t2*force_in[ei1]
__global__ void __launch_bounds__(128)
combine_scatter_kernel(const float* __restrict__ t1, const float* __restrict__ t2,
                       const float* __restrict__ dir,
                       const float* __restrict__ force_in,
                       float* __restrict__ force_out,
                       const int* __restrict__ ei0, const int* __restrict__ ei1) {
  const int e = blockIdx.x, f = threadIdx.x;
  const int i0 = ei0[e], i1 = ei1[e];
  const float a = t1[(size_t)e * F + f];
  const float b = t2[(size_t)e * F + f];
#pragma unroll
  for (int v = 0; v < 3; ++v) {
    float val = a * dir[e * 3 + v] + b * force_in[((size_t)i1 * 3 + v) * F + f];
    atomicAdd(&force_out[((size_t)i0 * 3 + v) * F + f], val);
  }
}

// atom[n] += sum_v force[n,v,:] * fw[n,v,:]
__global__ void __launch_bounds__(128)
atom_update_kernel(const float* __restrict__ force, const float* __restrict__ fw,
                   float* __restrict__ atom) {
  const int n = blockIdx.x, g = threadIdx.x;
  float s = 0.f;
#pragma unroll
  for (int v = 0; v < 3; ++v)
    s += force[((size_t)n * 3 + v) * F + g] * fw[((size_t)n * 3 + v) * F + g];
  atom[(size_t)n * F + g] += s;
}

__global__ void __launch_bounds__(128)
writeout_kernel(const float* __restrict__ atom, const float* __restrict__ force,
                float* __restrict__ out, int N) {
  const int n = blockIdx.x, f = threadIdx.x;
  out[(size_t)n * F + f] = atom[(size_t)n * F + f];
  const size_t base = (size_t)N * F;
#pragma unroll
  for (int v = 0; v < 3; ++v)
    out[base + ((size_t)n * 3 + v) * F + f] = force[((size_t)n * 3 + v) * F + f];
}

// ---------------------------------------------------------------------------
// Host orchestration
// ---------------------------------------------------------------------------
extern "C" void kernel_launch(void* const* d_in, const int* in_sizes, int n_in,
                              void* d_out, int out_size, void* d_ws, size_t ws_size,
                              hipStream_t stream) {
  (void)n_in; (void)out_size; (void)ws_size;
  const int N = in_sizes[0];
  const int E = in_sizes[1] / 3;

  const int*   z        = (const int*)  d_in[0];
  const float* disp     = (const float*)d_in[1];
  const int*   ei       = (const int*)  d_in[2];   // [2,E]
  const float* node_emb = (const float*)d_in[4];
  const float* mnp_W1   = (const float*)d_in[5];
  const float* mnp_b1   = (const float*)d_in[6];
  const float* mnp_W2   = (const float*)d_in[7];
  const float* mnp_b2   = (const float*)d_in[8];
  const float* me_W     = (const float*)d_in[9];
  const float* eq1_W1   = (const float*)d_in[10];
  const float* eq1_W2   = (const float*)d_in[11];
  const float* eq2_W1   = (const float*)d_in[12];
  const float* eq2_W2   = (const float*)d_in[13];
  const float* upd_W    = (const float*)d_in[14];
  const int* ei0 = ei;
  const int* ei1 = ei + E;

  // workspace carve (256B aligned)
  char* p = (char*)d_ws;
  auto carve = [&](size_t bytes) -> void* {
    void* r = (void*)p;
    p += (bytes + 255) & ~(size_t)255;
    return r;
  };
  const int WSZ = LAYERS * F * F;                  // one weight stack (3 layers)
  _Float16* whW1  = (_Float16*)carve((size_t)WSZ * 2);   // mnp_W1
  _Float16* whW2  = (_Float16*)carve((size_t)WSZ * 2);   // mnp_W2
  _Float16* wh11  = (_Float16*)carve((size_t)WSZ * 2);   // eq1_W1
  _Float16* wh12  = (_Float16*)carve((size_t)WSZ * 2);   // eq1_W2
  _Float16* wh21  = (_Float16*)carve((size_t)WSZ * 2);   // eq2_W1
  _Float16* wh22  = (_Float16*)carve((size_t)WSZ * 2);   // eq2_W2
  _Float16* whU   = (_Float16*)carve((size_t)WSZ * 2);   // upd_W
  float*    atom   = (float*)   carve((size_t)N * F * 4);
  float*    mnp    = (float*)   carve((size_t)N * F * 4);
  float*    forceA = (float*)   carve((size_t)N * 3 * F * 4);
  float*    forceB = (float*)   carve((size_t)N * 3 * F * 4);
  float*    fw     = (float*)   carve((size_t)N * 3 * F * 4);
  float*    dist_e = (float*)   carve((size_t)E * NB * 4);
  float*    dir_e  = (float*)   carve((size_t)E * 3 * 4);
  _Float16* msg    = (_Float16*)carve((size_t)E * F * 2);
  float*    t1     = (float*)   carve((size_t)E * F * 4);
  float*    t2     = (float*)   carve((size_t)E * F * 4);

  // --- weight f32 -> f16 conversion (once) ---
  {
    const int n = WSZ, blk = 256, grd = (n + blk - 1) / blk;
    f32_to_f16_kernel<<<grd, blk, 0, stream>>>(mnp_W1, whW1, n);
    f32_to_f16_kernel<<<grd, blk, 0, stream>>>(mnp_W2, whW2, n);
    f32_to_f16_kernel<<<grd, blk, 0, stream>>>(eq1_W1, wh11, n);
    f32_to_f16_kernel<<<grd, blk, 0, stream>>>(eq1_W2, wh12, n);
    f32_to_f16_kernel<<<grd, blk, 0, stream>>>(eq2_W1, wh21, n);
    f32_to_f16_kernel<<<grd, blk, 0, stream>>>(eq2_W2, wh22, n);
    f32_to_f16_kernel<<<grd, blk, 0, stream>>>(upd_W,  whU,  n);
  }

  // --- init atom / force, edge embedding ---
  init_kernel<<<N, F, 0, stream>>>(z, node_emb, atom, forceA);
  edge_embed_kernel<<<(E + 255) / 256, 256, 0, stream>>>(disp, dist_e, dir_e, E);

  float* fCur = forceA;
  float* fNxt = forceB;
  for (int l = 0; l < LAYERS; ++l) {
    const size_t wOfs = (size_t)l * F * F;
    // mnp = silu(atom @ W1^T + b1) @ W2^T + b2      (fp32 in, fused WMMA MLP)
    mlp2_wmma_kernel<float><<<N / 16, 256, 0, stream>>>(
        atom, whW1 + wOfs, mnp_b1 + (size_t)l * F,
        whW2 + wOfs, mnp_b2 + (size_t)l * F, mnp);
    // msg (f16) + atom scatter
    msg_kernel<<<E, F, 0, stream>>>(dist_e, me_W + (size_t)l * F * NB,
                                    mnp, ei0, ei1, msg, atom);
    // edge equivariant MLPs (f16 in, fused 2-GEMM, no bias)
    mlp2_wmma_kernel<_Float16><<<E / 16, 256, 0, stream>>>(
        msg, wh11 + wOfs, nullptr, wh12 + wOfs, nullptr, t1);
    mlp2_wmma_kernel<_Float16><<<E / 16, 256, 0, stream>>>(
        msg, wh21 + wOfs, nullptr, wh22 + wOfs, nullptr, t2);
    // force_next = force_cur ; += scatter(e1 + e2)
    hipMemcpyAsync(fNxt, fCur, (size_t)N * 3 * F * 4,
                   hipMemcpyDeviceToDevice, stream);
    combine_scatter_kernel<<<E, F, 0, stream>>>(t1, t2, dir_e, fCur, fNxt,
                                                ei0, ei1);
    // fw = force_next([3N,F]) @ upd_W^T ; atom += sum_v force*fw
    gemm_wmma_kernel<<<(3 * N) / 16, 256, 0, stream>>>(fNxt, whU + wOfs, fw);
    atom_update_kernel<<<N, F, 0, stream>>>(fNxt, fw, atom);
    // swap force buffers
    float* tmp = fCur; fCur = fNxt; fNxt = tmp;
  }

  writeout_kernel<<<N, F, 0, stream>>>(atom, fCur, (float*)d_out, N);
}
